// seg_model_pointnet2_59596966199815
// MI455X (gfx1250) — matrix operand order repaired and checked by reading, not compiled, not run on hardware
//
#include <hip/hip_runtime.h>
#include <hip/hip_bf16.h>
#include <float.h>

// ---------------------------------------------------------------------------
// Types for WMMA on gfx1250 (wave32).
// ---------------------------------------------------------------------------
typedef __attribute__((ext_vector_type(16))) _Float16 v16h;
typedef __attribute__((ext_vector_type(8)))  float    v8f;

struct alignas(16) Q16 { unsigned x, y, z, w; };
union V8F { v8f v; float f[8]; };

// A-matrix fragment (16x32 f16) from a row-major [16][Kpad] f16 buffer.
// ISA layout: lanes 0-15 hold K 0..7 (V0..3) and 16..23 (V4..7) of row M=lane;
// lanes 16-31 hold K 8..15 and 24..31 of row M=lane-16.
static __device__ __forceinline__ v16h load_afrag(const _Float16* actrow, int chunk, int lane) {
  const int kb = chunk * 32 + ((lane & 16) ? 8 : 0);
  union { v16h v; Q16 q[2]; } u;
  u.q[0] = *(const Q16*)(actrow + kb);
  u.q[1] = *(const Q16*)(actrow + kb + 16);
  return u.v;
}

// B-matrix fragment (32x16 f16) from row-major weights [out][Kpad] (B = W^T).
// ISA layout: lane n (0-15) holds K 0..7 and 16..23 of column n; lanes 16-31 hold K 8..15 and 24..31.
static __device__ __forceinline__ v16h load_bfrag(const _Float16* w, int Kpad, int tile, int chunk, int lane) {
  const int n  = lane & 15;
  const int kb = chunk * 32 + ((lane & 16) ? 8 : 0);
  const _Float16* p = w + (size_t)(tile * 16 + n) * Kpad + kb;
  union { v16h v; Q16 q[2]; } u;
  u.q[0] = *(const Q16*)(p);
  u.q[1] = *(const Q16*)(p + 16);
  return u.v;
}

static __device__ __forceinline__ v8f wmma_f16(v16h a, v16h b, v8f c) {
  return __builtin_amdgcn_wmma_f32_16x16x32_f16(false, a, false, b, (short)0, c, false, false);
}

// ---------------------------------------------------------------------------
// Constants
// ---------------------------------------------------------------------------
#define B_   8
#define N_   4096
#define K_   16
#define EPS  1e-5f

// ---------------------------------------------------------------------------
// Kernel 0: weight conversion to f16 (padded), BN folding, gf zeroing.
// ---------------------------------------------------------------------------
__global__ __launch_bounds__(256) void prep_kernel(
    const float* a0w, const float* a0b, const float* a0g, const float* a0t, const float* a0m, const float* a0v,
    const float* a1w, const float* a1b, const float* a1g, const float* a1t, const float* a1m, const float* a1v,
    const float* a2w, const float* a2b, const float* a2g, const float* a2t, const float* a2m, const float* a2v,
    const float* c2w, const float* c2b, const float* c2g, const float* c2t, const float* c2m, const float* c2v,
    const float* c3w, const float* c3b, const float* c3g, const float* c3t, const float* c3m, const float* c3v,
    const float* c4w, const float* c4b,
    _Float16* wa0, float* sb0, _Float16* wa1, float* sb1, _Float16* wa2, float* sb2,
    _Float16* wc2, float* sbc2, _Float16* wc3, float* sbc3, _Float16* wc4, float* bc4,
    float* gf)
{
  const int i = blockIdx.x * 256 + threadIdx.x; // < 65536
  // a0: (64,67) -> (64,96) padded
  if (i < 64 * 96)  { int o = i / 96, kk = i % 96; wa0[i] = (kk < 67) ? (_Float16)a0w[o * 67 + kk] : (_Float16)0.f; }
  if (i < 128 * 64) { wa1[i] = (_Float16)a1w[i]; }
  if (i < 128 * 128){ wa2[i] = (_Float16)a2w[i]; }
  if (i < 256 * 256){ wc2[i] = (_Float16)c2w[i]; }
  if (i < 128 * 256){ wc3[i] = (_Float16)c3w[i]; }
  // c4: (6,128) -> (16,128) padded rows
  if (i < 16 * 128) { int o = i / 128, kk = i % 128; wc4[i] = (o < 6) ? (_Float16)c4w[o * 128 + kk] : (_Float16)0.f; }
  if (i < B_ * 128) { gf[i] = 0.f; }
  // folded BN:  y = s*(Wx) + (s*b + t - s*m),  s = g*rsqrt(v+eps)
  if (i < 64)  { float s = a0g[i] * rsqrtf(a0v[i] + EPS); sb0[i]  = s; sb0[64 + i]   = s * a0b[i] + a0t[i] - s * a0m[i]; }
  if (i < 128) { float s = a1g[i] * rsqrtf(a1v[i] + EPS); sb1[i]  = s; sb1[128 + i]  = s * a1b[i] + a1t[i] - s * a1m[i]; }
  if (i < 128) { float s = a2g[i] * rsqrtf(a2v[i] + EPS); sb2[i]  = s; sb2[128 + i]  = s * a2b[i] + a2t[i] - s * a2m[i]; }
  if (i < 256) { float s = c2g[i] * rsqrtf(c2v[i] + EPS); sbc2[i] = s; sbc2[256 + i] = s * c2b[i] + c2t[i] - s * c2m[i]; }
  if (i < 128) { float s = c3g[i] * rsqrtf(c3v[i] + EPS); sbc3[i] = s; sbc3[128 + i] = s * c3b[i] + c3t[i] - s * c3m[i]; }
  if (i < 16)  { bc4[i] = (i < 6) ? c4b[i] : 0.f; }
}

// ---------------------------------------------------------------------------
// Kernel 1: per-point MLP 3 -> 64 -> 64 (BN+ReLU folded), output f16.
// ---------------------------------------------------------------------------
__global__ __launch_bounds__(256) void pointmlp_kernel(
    const float* __restrict__ points,
    const float* __restrict__ c0w, const float* __restrict__ c0b,
    const float* __restrict__ c0g, const float* __restrict__ c0t,
    const float* __restrict__ c0m, const float* __restrict__ c0v,
    const float* __restrict__ c1w, const float* __restrict__ c1b,
    const float* __restrict__ c1g, const float* __restrict__ c1t,
    const float* __restrict__ c1m, const float* __restrict__ c1v,
    _Float16* __restrict__ xf)
{
  const int p = blockIdx.x * 256 + threadIdx.x; // < B_*N_
  const float x = points[p * 3 + 0];
  const float y = points[p * 3 + 1];
  const float z = points[p * 3 + 2];
  float h[64];
#pragma unroll
  for (int o = 0; o < 64; ++o) {
    float a = c0w[o * 3 + 0] * x + c0w[o * 3 + 1] * y + c0w[o * 3 + 2] * z + c0b[o];
    float s = c0g[o] * rsqrtf(c0v[o] + EPS);
    h[o] = fmaxf(s * (a - c0m[o]) + c0t[o], 0.f);
  }
#pragma unroll 8
  for (int o = 0; o < 64; ++o) {
    float a = c1b[o];
#pragma unroll
    for (int i = 0; i < 64; ++i) a += c1w[o * 64 + i] * h[i];
    float s = c1g[o] * rsqrtf(c1v[o] + EPS);
    xf[(size_t)p * 64 + o] = (_Float16)fmaxf(s * (a - c1m[o]) + c1t[o], 0.f);
  }
}

// ---------------------------------------------------------------------------
// Kernel 2: kNN (k=16 smallest squared distances), register top-k scan.
// One thread = one query point; candidates tiled through LDS.
// ---------------------------------------------------------------------------
__global__ __launch_bounds__(256) void knn_kernel(const float* __restrict__ points,
                                                  int* __restrict__ idxbuf)
{
  __shared__ float sx[256], sy[256], sz[256];
  const int b  = blockIdx.x >> 4;
  const int qt = blockIdx.x & 15;
  const int q  = qt * 256 + threadIdx.x;
  const float* base = points + (size_t)b * N_ * 3;
  const float qx = base[q * 3 + 0], qy = base[q * 3 + 1], qz = base[q * 3 + 2];

  float bd[K_]; int bidx[K_];
#pragma unroll
  for (int s = 0; s < K_; ++s) { bd[s] = FLT_MAX; bidx[s] = 0; }

  for (int t = 0; t < N_ / 256; ++t) {
    const int c = t * 256 + threadIdx.x;
    sx[threadIdx.x] = base[c * 3 + 0];
    sy[threadIdx.x] = base[c * 3 + 1];
    sz[threadIdx.x] = base[c * 3 + 2];
    __syncthreads();
    for (int j = 0; j < 256; ++j) {
      const float dx = sx[j] - qx, dy = sy[j] - qy, dz = sz[j] - qz;
      const float d = dx * dx + dy * dy + dz * dz;
      if (d < bd[K_ - 1]) {
        float cd = d; int ci = t * 256 + j;
#pragma unroll
        for (int s = 0; s < K_; ++s) {
          if (cd < bd[s]) { float td = bd[s]; int ti = bidx[s]; bd[s] = cd; bidx[s] = ci; cd = td; ci = ti; }
        }
      }
    }
    __syncthreads();
  }
  int* op = idxbuf + ((size_t)b * N_ + q) * K_;
#pragma unroll
  for (int s = 0; s < K_; ++s) op[s] = bidx[s];
}

// ---------------------------------------------------------------------------
// Kernel 3: fused gather + per-neighbor MLP 67->64->128->128 + max over k.
// One wave = one point; its 16 neighbors are the 16 rows of the WMMA A tiles.
// ---------------------------------------------------------------------------
#define WAVES_C 8
__global__ __launch_bounds__(32 * WAVES_C) void group_mlp_max_kernel(
    const float* __restrict__ points, const _Float16* __restrict__ xf,
    const int* __restrict__ idxbuf,
    const _Float16* __restrict__ wa0, const float* __restrict__ sb0,
    const _Float16* __restrict__ wa1, const float* __restrict__ sb1,
    const _Float16* __restrict__ wa2, const float* __restrict__ sb2,
    float* __restrict__ localf, float* __restrict__ gf)
{
  __shared__ alignas(16) _Float16 act[WAVES_C][16][128]; // per-wave staging tile
  const int lane = threadIdx.x & 31;
  const int wv   = threadIdx.x >> 5;
  const int pt   = blockIdx.x * WAVES_C + wv; // global point id
  const int b    = pt >> 12;

  // ---- gather: rows = 16 neighbors; features [dxyz(3) | x64 | zeros->96]
  {
    const int j = lane & 15, half = lane >> 4;
    const int nb = idxbuf[(size_t)pt * K_ + j];
    const float* q = points + (size_t)pt * 3;
    const float* p = points + ((size_t)b * N_ + nb) * 3;
    const _Float16* xr = xf + ((size_t)b * N_ + nb) * 64;
    _Float16* row = &act[wv][j][0];
    if (half == 0) {
      row[0] = (_Float16)(p[0] - q[0]);
      row[1] = (_Float16)(p[1] - q[1]);
      row[2] = (_Float16)(p[2] - q[2]);
#pragma unroll
      for (int i = 0; i < 32; ++i) row[3 + i] = xr[i];
      for (int c = 67; c < 82; ++c) row[c] = (_Float16)0.f;
    } else {
#pragma unroll
      for (int i = 32; i < 64; ++i) row[3 + i] = xr[i];
      for (int c = 82; c < 96; ++c) row[c] = (_Float16)0.f;
    }
  }
  __syncthreads();

  const int nn = lane & 15;
  const int mb = (lane & 16) ? 8 : 0;
  const _Float16* myrow = &act[wv][nn][0];
  v16h av[4];

  // ---- a0: K=96 (3 chunks) -> 64 out (4 tiles)
#pragma unroll
  for (int c = 0; c < 3; ++c) av[c] = load_afrag(myrow, c, lane);
#pragma unroll
  for (int t = 0; t < 4; ++t) {
    V8F acc; acc.v = {};
#pragma unroll
    for (int c = 0; c < 3; ++c) acc.v = wmma_f16(av[c], load_bfrag(wa0, 96, t, c, lane), acc.v);
    const int ch = t * 16 + nn;
    const float s = sb0[ch], bi = sb0[64 + ch];
#pragma unroll
    for (int v = 0; v < 8; ++v)
      act[wv][mb + v][ch] = (_Float16)fmaxf(acc.f[v] * s + bi, 0.f);
  }
  __syncthreads();

  // ---- a1: K=64 (2 chunks) -> 128 out (8 tiles)
#pragma unroll
  for (int c = 0; c < 2; ++c) av[c] = load_afrag(myrow, c, lane);
#pragma unroll
  for (int t = 0; t < 8; ++t) {
    V8F acc; acc.v = {};
#pragma unroll
    for (int c = 0; c < 2; ++c) acc.v = wmma_f16(av[c], load_bfrag(wa1, 64, t, c, lane), acc.v);
    const int ch = t * 16 + nn;
    const float s = sb1[ch], bi = sb1[128 + ch];
#pragma unroll
    for (int v = 0; v < 8; ++v)
      act[wv][mb + v][ch] = (_Float16)fmaxf(acc.f[v] * s + bi, 0.f);
  }
  __syncthreads();

  // ---- a2: K=128 (4 chunks) -> 128 out (8 tiles) + max over 16 neighbors
#pragma unroll
  for (int c = 0; c < 4; ++c) av[c] = load_afrag(myrow, c, lane);
#pragma unroll
  for (int t = 0; t < 8; ++t) {
    V8F acc; acc.v = {};
#pragma unroll
    for (int c = 0; c < 4; ++c) acc.v = wmma_f16(av[c], load_bfrag(wa2, 128, t, c, lane), acc.v);
    const int ch = t * 16 + nn;
    const float s = sb2[ch], bi = sb2[128 + ch];
    float r = 0.f; // ReLU outputs are >= 0
#pragma unroll
    for (int v = 0; v < 8; ++v) r = fmaxf(r, fmaxf(acc.f[v] * s + bi, 0.f));
    r = fmaxf(r, __shfl_xor(r, 16, 32)); // merge the two M-halves
    if (lane < 16) {
      localf[(size_t)pt * 128 + ch] = r;
      atomicMax((int*)(gf + b * 128 + ch), __float_as_int(r)); // r >= 0: int cmp ok
    }
  }
}

// ---------------------------------------------------------------------------
// Kernel 4: head MLP 256 -> 256 -> 128 -> 6 (WMMA), one wave = 16 points.
// ---------------------------------------------------------------------------
#define WAVES_D 4
__global__ __launch_bounds__(32 * WAVES_D) void head_kernel(
    const float* __restrict__ localf, const float* __restrict__ gf,
    const _Float16* __restrict__ wc2, const float* __restrict__ sbc2,
    const _Float16* __restrict__ wc3, const float* __restrict__ sbc3,
    const _Float16* __restrict__ wc4, const float* __restrict__ bc4,
    float* __restrict__ out)
{
  __shared__ alignas(16) _Float16 f0[WAVES_D][16][256];
  __shared__ alignas(16) _Float16 f1[WAVES_D][16][256];
  const int lane = threadIdx.x & 31;
  const int wv   = threadIdx.x >> 5;
  const int tile = blockIdx.x * WAVES_D + wv; // 16-point tile id
  const int p0   = tile * 16;
  const int b    = p0 >> 12; // N_=4096 divisible by 16 -> one batch per tile

  // gather feat = [local(128) | gf(128)] as f16
  for (int e = lane; e < 16 * 256; e += 32) {
    const int r = e >> 8, c = e & 255;
    const float v = (c < 128) ? localf[(size_t)(p0 + r) * 128 + c] : gf[b * 128 + (c - 128)];
    f0[wv][r][c] = (_Float16)v;
  }
  __syncthreads();

  const int nn = lane & 15;
  const int mb = (lane & 16) ? 8 : 0;
  const _Float16* r0 = &f0[wv][nn][0];
  const _Float16* r1 = &f1[wv][nn][0];
  v16h av[8];

  // ---- c2: 256 -> 256
#pragma unroll
  for (int c = 0; c < 8; ++c) av[c] = load_afrag(r0, c, lane);
#pragma unroll
  for (int t = 0; t < 16; ++t) {
    V8F acc; acc.v = {};
#pragma unroll
    for (int c = 0; c < 8; ++c) acc.v = wmma_f16(av[c], load_bfrag(wc2, 256, t, c, lane), acc.v);
    const int ch = t * 16 + nn;
    const float s = sbc2[ch], bi = sbc2[256 + ch];
#pragma unroll
    for (int v = 0; v < 8; ++v)
      f1[wv][mb + v][ch] = (_Float16)fmaxf(acc.f[v] * s + bi, 0.f);
  }
  __syncthreads();

  // ---- c3: 256 -> 128
#pragma unroll
  for (int c = 0; c < 8; ++c) av[c] = load_afrag(r1, c, lane);
#pragma unroll
  for (int t = 0; t < 8; ++t) {
    V8F acc; acc.v = {};
#pragma unroll
    for (int c = 0; c < 8; ++c) acc.v = wmma_f16(av[c], load_bfrag(wc3, 256, t, c, lane), acc.v);
    const int ch = t * 16 + nn;
    const float s = sbc3[ch], bi = sbc3[128 + ch];
#pragma unroll
    for (int v = 0; v < 8; ++v)
      f0[wv][mb + v][ch] = (_Float16)fmaxf(acc.f[v] * s + bi, 0.f);
  }
  __syncthreads();

  // ---- c4: 128 -> 6 (padded to 16 out columns)
#pragma unroll
  for (int c = 0; c < 4; ++c) av[c] = load_afrag(r0, c, lane);
  V8F acc; acc.v = {};
#pragma unroll
  for (int c = 0; c < 4; ++c) acc.v = wmma_f16(av[c], load_bfrag(wc4, 128, 0, c, lane), acc.v);
  if (nn < 6) {
    const float bi = bc4[nn];
#pragma unroll
    for (int v = 0; v < 8; ++v)
      out[(size_t)(p0 + mb + v) * 6 + nn] = acc.f[v] + bi;
  }
}

// ---------------------------------------------------------------------------
// Launch
// ---------------------------------------------------------------------------
extern "C" void kernel_launch(void* const* d_in, const int* in_sizes, int n_in,
                              void* d_out, int out_size, void* d_ws, size_t ws_size,
                              hipStream_t stream) {
  (void)in_sizes; (void)n_in; (void)out_size; (void)ws_size;
  const float* points = (const float*)d_in[0];
  const float* c0w = (const float*)d_in[1];  const float* c0b = (const float*)d_in[2];
  const float* c0g = (const float*)d_in[3];  const float* c0t = (const float*)d_in[4];
  const float* c0m = (const float*)d_in[5];  const float* c0v = (const float*)d_in[6];
  const float* c1w = (const float*)d_in[7];  const float* c1b = (const float*)d_in[8];
  const float* c1g = (const float*)d_in[9];  const float* c1t = (const float*)d_in[10];
  const float* c1m = (const float*)d_in[11]; const float* c1v = (const float*)d_in[12];
  const float* a0w = (const float*)d_in[13]; const float* a0b = (const float*)d_in[14];
  const float* a0g = (const float*)d_in[15]; const float* a0t = (const float*)d_in[16];
  const float* a0m = (const float*)d_in[17]; const float* a0v = (const float*)d_in[18];
  const float* a1w = (const float*)d_in[19]; const float* a1b = (const float*)d_in[20];
  const float* a1g = (const float*)d_in[21]; const float* a1t = (const float*)d_in[22];
  const float* a1m = (const float*)d_in[23]; const float* a1v = (const float*)d_in[24];
  const float* a2w = (const float*)d_in[25]; const float* a2b = (const float*)d_in[26];
  const float* a2g = (const float*)d_in[27]; const float* a2t = (const float*)d_in[28];
  const float* a2m = (const float*)d_in[29]; const float* a2v = (const float*)d_in[30];
  const float* c2w = (const float*)d_in[31]; const float* c2b = (const float*)d_in[32];
  const float* c2g = (const float*)d_in[33]; const float* c2t = (const float*)d_in[34];
  const float* c2m = (const float*)d_in[35]; const float* c2v = (const float*)d_in[36];
  const float* c3w = (const float*)d_in[37]; const float* c3b = (const float*)d_in[38];
  const float* c3g = (const float*)d_in[39]; const float* c3t = (const float*)d_in[40];
  const float* c3m = (const float*)d_in[41]; const float* c3v = (const float*)d_in[42];
  const float* c4w = (const float*)d_in[43]; const float* c4b = (const float*)d_in[44];
  float* out = (float*)d_out;

  char* ws = (char*)d_ws;
  _Float16* xf     = (_Float16*)(ws + 0);         // 8*4096*64 f16   = 4 MiB
  int*      idxb   = (int*)     (ws + 4194304);   // 8*4096*16 i32   = 2 MiB
  float*    localf = (float*)   (ws + 6291456);   // 8*4096*128 f32  = 16 MiB
  float*    gf     = (float*)   (ws + 23068672);  // 8*128 f32
  _Float16* wa0    = (_Float16*)(ws + 23072768);  // 64*96 f16
  float*    sb0    = (float*)   (ws + 23085056);  // 2*64 f32
  _Float16* wa1    = (_Float16*)(ws + 23085568);  // 128*64 f16
  float*    sb1    = (float*)   (ws + 23101952);  // 2*128 f32
  _Float16* wa2    = (_Float16*)(ws + 23102976);  // 128*128 f16
  float*    sb2    = (float*)   (ws + 23135744);  // 2*128 f32
  _Float16* wc2    = (_Float16*)(ws + 23136768);  // 256*256 f16
  float*    sbc2   = (float*)   (ws + 23267840);  // 2*256 f32
  _Float16* wc3    = (_Float16*)(ws + 23269888);  // 128*256 f16
  float*    sbc3   = (float*)   (ws + 23335424);  // 2*128 f32
  _Float16* wc4    = (_Float16*)(ws + 23336448);  // 16*128 f16 (padded)
  float*    bc4    = (float*)   (ws + 23340544);  // 16 f32 (padded)

  prep_kernel<<<256, 256, 0, stream>>>(
      a0w, a0b, a0g, a0t, a0m, a0v, a1w, a1b, a1g, a1t, a1m, a1v,
      a2w, a2b, a2g, a2t, a2m, a2v, c2w, c2b, c2g, c2t, c2m, c2v,
      c3w, c3b, c3g, c3t, c3m, c3v, c4w, c4b,
      wa0, sb0, wa1, sb1, wa2, sb2, wc2, sbc2, wc3, sbc3, wc4, bc4, gf);

  pointmlp_kernel<<<(B_ * N_) / 256, 256, 0, stream>>>(
      points, c0w, c0b, c0g, c0t, c0m, c0v, c1w, c1b, c1g, c1t, c1m, c1v, xf);

  knn_kernel<<<B_ * (N_ / 256), 256, 0, stream>>>(points, idxb);

  group_mlp_max_kernel<<<(B_ * N_) / WAVES_C, 32 * WAVES_C, 0, stream>>>(
      points, xf, idxb, wa0, sb0, wa1, sb1, wa2, sb2, localf, gf);

  head_kernel<<<(B_ * N_ / 16) / WAVES_D, 32 * WAVES_D, 0, stream>>>(
      localf, gf, wc2, sbc2, wc3, sbc3, wc4, bc4, out);
}
